// GRUModel_7129645711670
// MI455X (gfx1250) — matrix operand order repaired and checked by reading, not compiled, not run on hardware
//
#include <hip/hip_runtime.h>
#include <hip/hip_bf16.h>
#include <math.h>

// GRU: B=64, S=512, I=128, H=1024, O=3
#define B_  64
#define S_  512
#define I_  128
#define H_  1024
#define O_  3
#define G3H (3 * H_)

// LDS row stride for staged w_hh rows: 1024 bf16 data + 8 bf16 pad
// (516 dwords == 4 mod 64 banks -> conflict-free ds_load_b128 across 16 rows)
#define LDSW 1032

typedef __attribute__((ext_vector_type(16))) __bf16 v16bf;
typedef __attribute__((ext_vector_type(8)))  float  v8f;
typedef unsigned short u16_t;

// ---------------- workspace layout (bytes) ----------------
#define WS_WHH   ((size_t)0)                       // 3H*H bf16 = 6,291,456 B
#define WS_WIH   (WS_WHH + (size_t)G3H * H_ * 2)   // 3H*I bf16 =   786,432 B
#define WS_HSEQ  (WS_WIH + (size_t)G3H * I_ * 2)   // (S+1)*B*H bf16 = 67,239,936 B

// ---------------- small helpers ----------------
__device__ __forceinline__ u16_t f2bf_u(float f) {
  union { float f; unsigned int u; } c; c.f = f;
  unsigned int u = c.u;
  u += 0x7FFFu + ((u >> 16) & 1u);        // round-to-nearest-even
  return (u16_t)(u >> 16);
}
__device__ __forceinline__ float bf2f(u16_t h) {
  union { float f; unsigned int u; } c; c.u = ((unsigned int)h) << 16; return c.f;
}
__device__ __forceinline__ float fast_sigmoid(float x) {
  return 1.0f / (1.0f + __expf(-x));
}
__device__ __forceinline__ float fast_tanh(float x) {
  x = fminf(fmaxf(x, -15.0f), 15.0f);
  float e = __expf(2.0f * x);
  return (e - 1.0f) / (e + 1.0f);
}

// A-fragment (16x32 bf16, MxK): lane l -> row M=l%16; kb = 16*(l>=16);
// elements 0..15 = K kb..kb+15 (contiguous along K).
__device__ __forceinline__ v16bf load_frag_a(const u16_t* p) {
  union { uint4 q[2]; v16bf v; } u;
  u.q[0] = *(const uint4*)(p);
  u.q[1] = *(const uint4*)(p + 8);
  return u.v;
}
// B-fragment (32x16 bf16, KxN): lane l -> col N=l%16; kb = 16*(l>=16);
// elements 0..15 = K kb..kb+15 (contiguous along K).
__device__ __forceinline__ v16bf load_frag_b(const u16_t* p) {
  union { uint4 q[2]; v16bf v; } u;
  u.q[0] = *(const uint4*)(p);
  u.q[1] = *(const uint4*)(p + 8);
  return u.v;
}
// A-fragment built from fp32 source (on-the-fly bf16 conversion of u_t)
__device__ __forceinline__ v16bf cvt_frag_a(const float* p) {
  union { u16_t s[16]; v16bf v; } u;
  #pragma unroll
  for (int i = 0; i < 8; ++i) {
    u.s[i]     = f2bf_u(p[i]);
    u.s[8 + i] = f2bf_u(p[8 + i]);
  }
  return u.v;
}

#define WMMA_BF16(A, Bm, C) \
  __builtin_amdgcn_wmma_f32_16x16x32_bf16(false, (A), false, (Bm), (short)0, (C), false, false)

// ---------------- utility kernels ----------------
__global__ void cvt_bf16_kernel(const float* __restrict__ src,
                                u16_t* __restrict__ dst, int n) {
  int i = blockIdx.x * blockDim.x + threadIdx.x;
  if (i < n) dst[i] = f2bf_u(src[i]);
}

__global__ void zero_bf16_kernel(u16_t* __restrict__ dst, int n) {
  int i = blockIdx.x * blockDim.x + threadIdx.x;
  if (i < n) dst[i] = 0;
}

// ---------------- GRU recurrent step ----------------
// Grid: H/16 = 64 blocks (one per 16-column tile of H), block = 128 threads
// (4 waves = 4 row tiles of B).  The 48 w_hh rows this column tile needs
// (r/z/n gates) are staged into LDS once per block via gfx1250 async
// global->LDS copies (ASYNCcnt), then all 4 waves stream B-fragments from
// LDS (conflict-free via row padding) while A-fragments come from global.
__global__ __launch_bounds__(128)
void gru_step_kernel(const float* __restrict__ u,
                     const u16_t* __restrict__ wih,   // [3H, I] bf16 row-major
                     const u16_t* __restrict__ whh,   // [3H, H] bf16 row-major
                     const float* __restrict__ b_ih,
                     const float* __restrict__ b_hh,
                     u16_t* __restrict__ hseq,        // (S+1) slots of [B, H] bf16
                     int t)
{
  __shared__ u16_t lds_whh[48 * LDSW];   // 99,072 B

  const int tid  = threadIdx.x;
  const int lane = tid & 31;
  const int rt   = tid >> 5;       // wave id = row tile of B
  const int nl = lane & 15;        // N / M-within-tile index
  const int lh = lane >> 4;        // lane-half selector
  const int j0 = blockIdx.x * 16;  // column tile of H
  const int j  = j0 + nl;
  const int arow = rt * 16 + nl;   // batch row this lane's A fragment covers

  const u16_t* hprev = hseq + (size_t)t       * (B_ * H_);
  u16_t*       hnext = hseq + (size_t)(t + 1) * (B_ * H_);

  // ---- async-stage the 48 w_hh rows (gates r,z,n x 16 cols) into LDS ----
  // 48 rows x 2048 B = 6144 16-byte segments, spread over 128 threads.
  for (int i = tid; i < 48 * 128; i += 128) {
    const int r   = i >> 7;          // staged row 0..47 (gate*16 + n)
    const int seg = i & 127;         // 16B segment within the row
    const int grow = (r >> 4) * H_ + j0 + (r & 15);
    const u16_t* gp = whh + (size_t)grow * H_ + seg * 8;
    const unsigned lp = (unsigned)(uintptr_t)&lds_whh[r * LDSW + seg * 8];
    asm volatile("global_load_async_to_lds_b128 %0, %1, off"
                 :: "v"(lp), "v"((unsigned long long)(uintptr_t)gp)
                 : "memory");
  }
  asm volatile("s_wait_asynccnt 0x0" ::: "memory");
  __syncthreads();

  v8f accr = {}, accz = {}, acchn = {}, accxn = {};

  // ---- hidden GEMM: h_prev [B,H] x w_hh^T [H,3H], K = H = 1024 ----
  const u16_t* ap  = hprev + (size_t)arow * H_ + lh * 16;
  const u16_t* lbr = &lds_whh[(0 * 16 + nl) * LDSW + lh * 16];
  const u16_t* lbz = &lds_whh[(1 * 16 + nl) * LDSW + lh * 16];
  const u16_t* lbn = &lds_whh[(2 * 16 + nl) * LDSW + lh * 16];

  #pragma unroll 4
  for (int kc = 0; kc < H_; kc += 32) {
    v16bf a  = load_frag_a(ap + kc);
    v16bf br = load_frag_b(lbr + kc);
    v16bf bz = load_frag_b(lbz + kc);
    v16bf bn = load_frag_b(lbn + kc);
    accr  = WMMA_BF16(a, br, accr);
    accz  = WMMA_BF16(a, bz, accz);
    acchn = WMMA_BF16(a, bn, acchn);
  }

  // ---- input GEMM (fused): u_t [B,I] x w_ih^T [I,3H], K = I = 128 ----
  const float* up  = u + ((size_t)arow * S_ + t) * I_ + lh * 16;
  const u16_t* cpr = wih + (size_t)(j)          * I_ + lh * 16;
  const u16_t* cpz = wih + (size_t)(H_ + j)     * I_ + lh * 16;
  const u16_t* cpn = wih + (size_t)(2 * H_ + j) * I_ + lh * 16;

  #pragma unroll
  for (int kc = 0; kc < I_; kc += 32) {
    v16bf a  = cvt_frag_a(up + kc);
    v16bf br = load_frag_b(cpr + kc);
    v16bf bz = load_frag_b(cpz + kc);
    v16bf bn = load_frag_b(cpn + kc);
    accr  = WMMA_BF16(a, br, accr);
    accz  = WMMA_BF16(a, bz, accz);
    accxn = WMMA_BF16(a, bn, accxn);
  }

  // ---- gated epilogue; C/D layout: VGPR v -> M = v + 8*lh, N = lane%16 ----
  const float br_b = b_ih[j]          + b_hh[j];
  const float bz_b = b_ih[H_ + j]     + b_hh[H_ + j];
  const float bxn  = b_ih[2 * H_ + j];
  const float bhn  = b_hh[2 * H_ + j];

  #pragma unroll
  for (int v = 0; v < 8; ++v) {
    const int row = rt * 16 + v + 8 * lh;
    const float hp = bf2f(hprev[(size_t)row * H_ + j]);
    const float r  = fast_sigmoid(accr[v] + br_b);
    const float z  = fast_sigmoid(accz[v] + bz_b);
    const float n  = fast_tanh(accxn[v] + bxn + r * (acchn[v] + bhn));
    const float hn = (1.0f - z) * n + z * hp;
    hnext[(size_t)row * H_ + j] = f2bf_u(hn);
  }
}

// ---------------- output projection: out[b,s,:] = h_s[b,:] @ w_fc^T + b_fc ----
__global__ __launch_bounds__(32)
void gru_proj_kernel(const u16_t* __restrict__ hseq,
                     const float* __restrict__ w_fc,  // [O, H]
                     const float* __restrict__ b_fc,  // [O]
                     float* __restrict__ out)         // [B, S, O]
{
  const int bs = blockIdx.x;           // b * S + s
  const int b = bs / S_;
  const int s = bs - b * S_;
  const int lane = threadIdx.x;

  const u16_t* hrow = hseq + ((size_t)(s + 1) * B_ + b) * H_;

  float a0 = 0.0f, a1 = 0.0f, a2 = 0.0f;
  for (int k = lane; k < H_; k += 32) {
    const float hv = bf2f(hrow[k]);
    a0 += hv * w_fc[k];
    a1 += hv * w_fc[H_ + k];
    a2 += hv * w_fc[2 * H_ + k];
  }
  #pragma unroll
  for (int off = 16; off > 0; off >>= 1) {
    a0 += __shfl_xor(a0, off, 32);
    a1 += __shfl_xor(a1, off, 32);
    a2 += __shfl_xor(a2, off, 32);
  }
  if (lane == 0) {
    float* o = out + (size_t)bs * O_;
    o[0] = a0 + b_fc[0];
    o[1] = a1 + b_fc[1];
    o[2] = a2 + b_fc[2];
  }
}

// ---------------- host launcher ----------------
extern "C" void kernel_launch(void* const* d_in, const int* in_sizes, int n_in,
                              void* d_out, int out_size, void* d_ws, size_t ws_size,
                              hipStream_t stream) {
  (void)in_sizes; (void)n_in; (void)out_size; (void)ws_size;

  const float* u    = (const float*)d_in[0];   // [B, S, I]
  const float* w_ih = (const float*)d_in[1];   // [3H, I]
  const float* w_hh = (const float*)d_in[2];   // [3H, H]
  const float* b_ih = (const float*)d_in[3];   // [3H]
  const float* b_hh = (const float*)d_in[4];   // [3H]
  const float* w_fc = (const float*)d_in[5];   // [O, H]
  const float* b_fc = (const float*)d_in[6];   // [O]
  float* out = (float*)d_out;                  // [B, S, O]

  char* ws = (char*)d_ws;
  u16_t* whh_bf = (u16_t*)(ws + WS_WHH);
  u16_t* wih_bf = (u16_t*)(ws + WS_WIH);
  u16_t* hseq   = (u16_t*)(ws + WS_HSEQ);

  // per-launch weight conversion + h_0 = 0 (deterministic, no cached state)
  { int n = G3H * H_; cvt_bf16_kernel<<<(n + 255) / 256, 256, 0, stream>>>(w_hh, whh_bf, n); }
  { int n = G3H * I_; cvt_bf16_kernel<<<(n + 255) / 256, 256, 0, stream>>>(w_ih, wih_bf, n); }
  { int n = B_ * H_;  zero_bf16_kernel<<<(n + 255) / 256, 256, 0, stream>>>(hseq, n); }

  for (int t = 0; t < S_; ++t) {
    gru_step_kernel<<<H_ / 16, 128, 0, stream>>>(u, wih_bf, whh_bf, b_ih, b_hh, hseq, t);
  }

  gru_proj_kernel<<<B_ * S_, 32, 0, stream>>>(hseq, w_fc, b_fc, out);
}
